// AlignmentLoss_26706106646932
// MI455X (gfx1250) — matrix operand order repaired and checked by reading, not compiled, not run on hardware
//
#include <hip/hip_runtime.h>
#include <stdint.h>

typedef _Float16 v16h __attribute__((ext_vector_type(16)));
typedef float    v8f  __attribute__((ext_vector_type(8)));

#define Bsz 8
#define Hh  96
#define Ww  96
#define Dd  768
#define Nn  (Hh*Ww)          // 9216
#define KC  32               // f16 K elements per chunk
#define NCH (Dd/KC)          // 24
#define TH  2                // rows of z1 per block
#define Z2R (TH+4)           // 6 z2 rows (dy halo)
#define Z2C 112              // 96 cols + dx halo + B-tile overrun pad
#define NBLK (Bsz*(Hh/TH))   // 384 main blocks

// ---------------- async global -> LDS (CDNA5) ----------------
__device__ __forceinline__ void g2l_b128(void* lds, const void* g) {
  uint32_t l = (uint32_t)(size_t)lds;   // low 32 bits of flat LDS addr = LDS offset
  asm volatile("global_load_async_to_lds_b128 %0, %1, off"
               :: "v"(l), "v"(g) : "memory");
}
__device__ __forceinline__ void wait_async0() {
  asm volatile("s_wait_asynccnt 0x0" ::: "memory");
}

// load 32 bytes (16 f16) from two 16B LDS locations into a WMMA operand
__device__ __forceinline__ v16h ld32(const _Float16* p0, const _Float16* p1) {
  union { uint4 u[2]; v16h h; } t;
  t.u[0] = *(const uint4*)p0;
  t.u[1] = *(const uint4*)p1;
  return t.h;
}

// ---------------- kernel 1: per-batch weight sums ----------------
__global__ void k_wsum(const float* __restrict__ w1, const float* __restrict__ w2,
                       float* __restrict__ sums) {
  int bid = blockIdx.x;                 // 0..15
  const float* w = (bid < 8) ? w1 : w2;
  int b = bid & 7;
  float s = 0.f;
  for (int i = threadIdx.x; i < Nn; i += 256) s += w[(size_t)b*Nn + i];
  #pragma unroll
  for (int off = 16; off > 0; off >>= 1) s += __shfl_xor(s, off, 32);
  __shared__ float r[8];
  int wv = threadIdx.x >> 5, lane = threadIdx.x & 31;
  if (lane == 0) r[wv] = s;
  __syncthreads();
  if (threadIdx.x == 0) {
    float t = 0.f;
    #pragma unroll
    for (int i = 0; i < 8; i++) t += r[i];
    sums[bid] = t;
  }
}

// ---------------- kernel 2: normalize + f16 convert ----------------
// one wave per pixel; 2*B*N waves total, 8 waves per 256-thread block
__global__ void k_norm(const float* __restrict__ z1, const float* __restrict__ z2,
                       _Float16* __restrict__ z1h, _Float16* __restrict__ z2h) {
  int wid = blockIdx.x * 8 + (threadIdx.x >> 5);
  int lane = threadIdx.x & 31;
  const float* src; _Float16* dst; int pix;
  if (wid < Bsz*Nn) { src = z1; dst = z1h; pix = wid; }
  else              { src = z2; dst = z2h; pix = wid - Bsz*Nn; }
  size_t base = (size_t)pix * Dd;
  float4 v[6];
  float ssq = 0.f;
  #pragma unroll
  for (int i = 0; i < 6; i++) {
    v[i] = *(const float4*)(src + base + lane*4 + i*128);
    ssq += v[i].x*v[i].x + v[i].y*v[i].y + v[i].z*v[i].z + v[i].w*v[i].w;
  }
  #pragma unroll
  for (int off = 16; off > 0; off >>= 1) ssq += __shfl_xor(ssq, off, 32);
  float inv = 1.0f / fmaxf(sqrtf(ssq), 1e-12f);
  #pragma unroll
  for (int i = 0; i < 6; i++) {
    _Float16 h[4];
    h[0] = (_Float16)(v[i].x * inv);
    h[1] = (_Float16)(v[i].y * inv);
    h[2] = (_Float16)(v[i].z * inv);
    h[3] = (_Float16)(v[i].w * inv);
    *(uint2*)(dst + base + lane*4 + i*128) = *(uint2*)h;
  }
}

// ---------------- kernel 3: shifted correlation via WMMA + softmax ----------------
// grid: 384 blocks (8 batches x 48 row-blocks of TH=2 rows), 192 threads (6 waves)
// wave wv: row ri = wv/3, 32-col third mq = wv%3 -> two 16-px M-tiles
// Double-buffered async global->LDS staging overlapped with WMMA compute.
__global__ void __launch_bounds__(192)
k_main(const _Float16* __restrict__ z1h, const _Float16* __restrict__ z2h,
       const float* __restrict__ w1, const float* __restrict__ w2,
       const float* __restrict__ sums, float* __restrict__ part) {
  __shared__ union __align__(16) {
    struct {
      _Float16 z1t[2][TH*Ww*KC];     // 2 x 12 KB   [buf][local pixel][k]
      _Float16 z2t[2][Z2R*Z2C*KC];   // 2 x 43 KB   [buf][row][col+2][k]
    } s;
    float sims[TH*Ww*25];            // 19.2 KB, overlays tiles after K loop
  } u;
  __shared__ float red[16];

  const int b  = blockIdx.x / (Hh/TH);
  const int rb = blockIdx.x % (Hh/TH);
  const int r0 = rb * TH;

  // zero both z2 tile buffers once (borders / out-of-image cells stay zero)
  for (int i = threadIdx.x; i < (2*Z2R*Z2C*KC)/2; i += 192)
    ((uint32_t*)u.s.z2t)[i] = 0u;

  const int wv = threadIdx.x >> 5, lane = threadIdx.x & 31;
  const int lo = lane & 15, hi = lane >> 4;
  const int ri = wv / 3, mq = wv - ri*3;

  // async-stage one K-chunk into buffer bi
  auto stage = [&](int bi, int kc) {
    // z1: TH*96 pixels * 64B, 16B per async op (4 iters/thread)
    for (int i = threadIdx.x; i < TH*Ww*4; i += 192) {
      int px = i >> 2, sub = i & 3;
      int row = px / Ww, col = px - row*Ww;
      size_t gp = ((size_t)(b*Nn + (r0+row)*Ww + col))*Dd + kc*KC + sub*8;
      g2l_b128(u.s.z1t[bi] + px*KC + sub*8, z1h + gp);
    }
    // z2: Z2R rows with dy halo; skip out-of-image rows (12 iters/thread)
    for (int i = threadIdx.x; i < Z2R*Ww*4; i += 192) {
      int px = i >> 2, sub = i & 3;
      int lr = px / Ww, col = px - lr*Ww;
      int rr = r0 - 2 + lr;
      if (rr >= 0 && rr < Hh) {
        size_t gp = ((size_t)(b*Nn + rr*Ww + col))*Dd + kc*KC + sub*8;
        g2l_b128(u.s.z2t[bi] + (lr*Z2C + col + 2)*KC + sub*8, z2h + gp);
      }
    }
  };

  v8f acc[2][5][2];
  #pragma unroll
  for (int mt = 0; mt < 2; mt++)
    #pragma unroll
    for (int dy = 0; dy < 5; dy++)
      #pragma unroll
      for (int t = 0; t < 2; t++)
        acc[mt][dy][t] = (v8f){0.f,0.f,0.f,0.f,0.f,0.f,0.f,0.f};

  __syncthreads();          // zero-fill visible everywhere
  stage(0, 0);              // preload first chunk

  for (int kc = 0; kc < NCH; ++kc) {
    const int cur = kc & 1;
    wait_async0();          // this wave's chunk-kc loads complete
    __syncthreads();        // all waves' chunk-kc data in LDS; compute kc-1 done

    // prefetch next chunk into the other buffer (overlaps with WMMA below)
    if (kc + 1 < NCH) stage(1 - cur, kc + 1);

    // ---- hoist ALL operand loads: 2 A tiles + 15 B tiles live at once ----
    v16h A[2];
    v16h Bm[5][3];
    #pragma unroll
    for (int mt = 0; mt < 2; mt++) {
      const _Float16* p = u.s.z1t[cur] + (ri*Ww + mq*32 + mt*16 + lo)*KC + hi*8;
      A[mt] = ld32(p, p + 16);
    }
    #pragma unroll
    for (int dy = 0; dy < 5; dy++)
      #pragma unroll
      for (int j = 0; j < 3; j++) {
        const _Float16* p =
            u.s.z2t[cur] + ((ri+dy)*Z2C + mq*32 + 16*j + lo)*KC + hi*8;
        Bm[dy][j] = ld32(p, p + 16);
      }

    // ---- 20 banded WMMAs, loads already in flight/registers ----
    #pragma unroll
    for (int dy = 0; dy < 5; dy++)
      #pragma unroll
      for (int mt = 0; mt < 2; mt++)
        #pragma unroll
        for (int t = 0; t < 2; t++)
          acc[mt][dy][t] = __builtin_amdgcn_wmma_f32_16x16x32_f16(
              false, A[mt], false, Bm[dy][mt+t], (short)0, acc[mt][dy][t],
              false, false);
  }

  // ---- extract 5-wide diagonal band -> sims[pixel][25] (overlays tiles) ----
  __syncthreads();
  #pragma unroll
  for (int mt = 0; mt < 2; mt++)
    #pragma unroll
    for (int dy = 0; dy < 5; dy++)
      #pragma unroll
      for (int t = 0; t < 2; t++)
        #pragma unroll
        for (int v = 0; v < 8; v++) {
          int m  = v + 8*hi;                 // row within M-tile (C/D layout)
          int dx = t*16 + lo - 2 - m;        // column offset - row -> shift
          if (dx >= -2 && dx <= 2) {
            int pixl = ri*Ww + mq*32 + mt*16 + m;
            u.sims[pixl*25 + dy*5 + dx + 2] = acc[mt][dy][t][v];
          }
        }
  __syncthreads();

  // ---- per-pixel softmax over 25 shifts + weighted partials ----
  {
    int t0 = threadIdx.x;                    // pixel 0..191 in this block
    float sv[25];
    #pragma unroll
    for (int i = 0; i < 25; i++) sv[i] = u.sims[t0*25 + i];
    float mx = sv[0];
    #pragma unroll
    for (int i = 1; i < 25; i++) mx = fmaxf(mx, sv[i]);
    const float invT = 1.0f / (0.07f + 1e-10f);
    float den = 0.f, num = 0.f;
    #pragma unroll
    for (int i = 0; i < 25; i++) {
      float e = __expf((sv[i] - mx) * invT);
      den += e; num += e * sv[i];
    }
    float sb = num / den;

    size_t pg = (size_t)b*Nn + (size_t)r0*Ww + t0;
    float wa = 0.5f * (w1[pg] / (sums[b] + 1e-10f) +
                       w2[pg] / (sums[8 + b] + 1e-10f));
    float la = (1.0f - sb) * wa;

    #pragma unroll
    for (int off = 16; off > 0; off >>= 1) {
      la += __shfl_xor(la, off, 32);
      sb += __shfl_xor(sb, off, 32);
    }
    if (lane == 0) { red[wv] = la; red[8 + wv] = sb; }
    __syncthreads();
    if (threadIdx.x == 0) {
      float a = 0.f, p = 0.f;
      #pragma unroll
      for (int w = 0; w < 6; w++) { a += red[w]; p += red[8 + w]; }
      part[blockIdx.x*2 + 0] = a;
      part[blockIdx.x*2 + 1] = p;
    }
  }
}

// ---------------- kernel 4: deterministic final reduction ----------------
__global__ void k_final(const float* __restrict__ part, float* __restrict__ out) {
  __shared__ float ra[256], rp[256];
  float a = 0.f, p = 0.f;
  for (int i = threadIdx.x; i < NBLK; i += 256) {
    a += part[2*i + 0];
    p += part[2*i + 1];
  }
  ra[threadIdx.x] = a; rp[threadIdx.x] = p;
  __syncthreads();
  for (int off = 128; off > 0; off >>= 1) {
    if (threadIdx.x < off) {
      ra[threadIdx.x] += ra[threadIdx.x + off];
      rp[threadIdx.x] += rp[threadIdx.x + off];
    }
    __syncthreads();
  }
  if (threadIdx.x == 0) {
    out[0] = ra[0] / (float)Bsz;             // align_loss
    out[1] = rp[0] / (float)(Bsz*Nn);        // pos_similarity
  }
}

// ---------------- host launcher ----------------
extern "C" void kernel_launch(void* const* d_in, const int* in_sizes, int n_in,
                              void* d_out, int out_size, void* d_ws, size_t ws_size,
                              hipStream_t stream) {
  const float* z1 = (const float*)d_in[0];
  const float* z2 = (const float*)d_in[1];
  const float* w1 = (const float*)d_in[2];
  const float* w2 = (const float*)d_in[3];
  (void)in_sizes; (void)n_in; (void)out_size; (void)ws_size;

  char* ws = (char*)d_ws;
  const size_t halfBytes = (size_t)Bsz * Nn * Dd * sizeof(_Float16); // 113,246,208
  _Float16* z1h = (_Float16*)ws;
  _Float16* z2h = (_Float16*)(ws + halfBytes);
  float* sums   = (float*)(ws + 2*halfBytes);   // 16 floats
  float* part   = sums + 16;                    // 2*NBLK floats
  float* out    = (float*)d_out;

  k_wsum <<<dim3(16),               dim3(256), 0, stream>>>(w1, w2, sums);
  k_norm <<<dim3(2*Bsz*Nn/8),       dim3(256), 0, stream>>>(z1, z2, z1h, z2h);
  k_main <<<dim3(NBLK),             dim3(192), 0, stream>>>(z1h, z2h, w1, w2, sums, part);
  k_final<<<dim3(1),                dim3(256), 0, stream>>>(part, out);
}
